// PointNet2_SSN_87591563035217
// MI455X (gfx1250) — matrix operand order repaired and checked by reading, not compiled, not run on hardware
//
#include <hip/hip_runtime.h>

// ---------------------------------------------------------------------------
// PointNet++ part-seg forward for MI455X (gfx1250, wave32, WMMA).
// All MLP layers run as f16 WMMA GEMMs (v_wmma_f32_16x16x32_f16) with f32
// accumulate; each wave register-blocks a 16x64 output strip (4 N-tiles) to
// reuse the A fragment 4x. K loop is unrolled by 2 with two short-lived
// fragment sets so loads overlap WMMAs without register spills.
// ---------------------------------------------------------------------------

typedef _Float16 v16h __attribute__((ext_vector_type(16)));
typedef _Float16 v8h  __attribute__((ext_vector_type(8)));
typedef float    v8f  __attribute__((ext_vector_type(8)));

#define BATCH 16
#define NPTS  8192

union Frag { v16h v; v8h h[2]; };

__device__ __forceinline__ void load_frags(const _Float16* __restrict__ arow,
                                           const _Float16* __restrict__ brow,
                                           size_t btile, int k0,
                                           Frag& a, Frag (&b)[4]) {
  a.h[0] = *(const v8h*)(arow + k0);
  a.h[1] = *(const v8h*)(arow + k0 + 16);
#pragma unroll
  for (int j = 0; j < 4; ++j) {
    b[j].h[0] = *(const v8h*)(brow + (size_t)j * btile + k0);
    b[j].h[1] = *(const v8h*)(brow + (size_t)j * btile + k0 + 16);
  }
}

__device__ __forceinline__ void wmma4(const Frag& a, const Frag (&b)[4],
                                      v8f (&acc)[4]) {
#pragma unroll
  for (int j = 0; j < 4; ++j)
    acc[j] = __builtin_amdgcn_wmma_f32_16x16x32_f16(
        false, a.v, false, b[j].v, (short)0, acc[j], false, false);
}

// ---------------------------------------------------------------------------
// Weight conversion: f32 [O x K] -> f16 [Op x Kp], zero padded.
// ---------------------------------------------------------------------------
__global__ void convert_w(const float* __restrict__ W, int O, int K,
                          _Float16* __restrict__ w16, int Op, int Kp) {
  int idx = blockIdx.x * blockDim.x + threadIdx.x;
  if (idx >= Op * Kp) return;
  int o = idx / Kp, k = idx % Kp;
  float v = (o < O && k < K) ? W[(size_t)o * K + k] : 0.f;
  w16[idx] = (_Float16)v;
}

// ---------------------------------------------------------------------------
// WMMA GEMM: D[M x O] = A[M x Kp(f16)] * Wt[Op x Kp(f16)]^T + bias.
// One 16x64 strip (4 N-tiles) per wave; K loop unrolled by 2 (two fragment
// sets per iteration, each loaded then consumed -> short live ranges).
// A fragment (16-bit A 16x32): lane l: m = l&15, halves = K[khalf*8..+7],
//   then K[16+khalf*8..+7] where khalf = l>>4.   (two b128 loads)
// B fragment (16-bit B 32x16): lane l: n = l&15, halves = K[khalf*8..+7],
//   then K[16+khalf*8..+7] (same k order as A).
// C layout: VGPR r, lane l -> M = r + 8*(l>>4), N = l&15.
// ---------------------------------------------------------------------------
__global__ void wmma_gemm(const _Float16* __restrict__ A, int ldA,
                          const _Float16* __restrict__ Wm,
                          const float* __restrict__ bias,
                          int M, int O, int Kp,
                          _Float16* __restrict__ out16, int ld16,
                          float* __restrict__ out32, int ldO) {
  const int waveId = threadIdx.x >> 5;
  const int lane   = threadIdx.x & 31;
  const int mTile  = blockIdx.x * 4 + waveId;
  if (mTile * 16 >= M) return;           // wave-uniform
  const int nBase  = blockIdx.y * 64;

  const int mlo   = lane & 15;
  const int khalf = lane >> 4;
  const _Float16* arow = A  + (size_t)(mTile * 16 + mlo) * ldA + khalf * 8;
  const _Float16* brow = Wm + (size_t)(nBase + mlo) * Kp  + khalf * 8;
  const size_t btile = (size_t)16 * Kp;  // halves between adjacent N tiles

  v8f acc[4];
#pragma unroll
  for (int j = 0; j < 4; ++j) acc[j] = (v8f){0.f,0.f,0.f,0.f,0.f,0.f,0.f,0.f};

  int k = 0;
  for (; k + 64 <= Kp; k += 64) {
    Frag a0, b0[4], a1, b1[4];
    load_frags(arow, brow, btile, k,      a0, b0);
    load_frags(arow, brow, btile, k + 32, a1, b1);
    wmma4(a0, b0, acc);
    wmma4(a1, b1, acc);
  }
  if (k < Kp) {
    Frag a0, b0[4];
    load_frags(arow, brow, btile, k, a0, b0);
    wmma4(a0, b0, acc);
  }

  const int mbase = mTile * 16 + khalf * 8;
#pragma unroll
  for (int j = 0; j < 4; ++j) {
    const int n = nBase + j * 16 + mlo;
    const float bv = (n < O) ? bias[n] : 0.f;
#pragma unroll
    for (int r = 0; r < 8; ++r) {
      float v = acc[j][r] + bv;
      size_t m = (size_t)(mbase + r);
      if (out32) {
        if (n < O) out32[m * ldO + n] = v;
      } else {
        out16[m * ld16 + n] = (n < O) ? (_Float16)v : (_Float16)0.f;
      }
    }
  }
}

// ---------------------------------------------------------------------------
// Batch-norm statistics (sum / sumsq per channel) + apply (+ReLU), f16 inplace.
// ---------------------------------------------------------------------------
__global__ void bn_stats(const _Float16* __restrict__ x, int O, int M,
                         float* __restrict__ stats) {
  const int c = blockIdx.x;
  const int r0 = blockIdx.y * 4096;
  const int r1 = min(r0 + 4096, M);
  float s = 0.f, ss = 0.f;
  for (int r = r0 + threadIdx.x; r < r1; r += 256) {
    float v = (float)x[(size_t)r * O + c];
    s += v; ss += v * v;
  }
  __shared__ float shs[256], shq[256];
  shs[threadIdx.x] = s; shq[threadIdx.x] = ss;
  __syncthreads();
  for (int st = 128; st > 0; st >>= 1) {
    if (threadIdx.x < st) { shs[threadIdx.x] += shs[threadIdx.x + st];
                            shq[threadIdx.x] += shq[threadIdx.x + st]; }
    __syncthreads();
  }
  if (threadIdx.x == 0) {
    atomicAdd(&stats[c], shs[0]);
    atomicAdd(&stats[O + c], shq[0]);
  }
}

__global__ void bn_apply(_Float16* __restrict__ x, int M, int O,
                         const float* __restrict__ stats,
                         const float* __restrict__ gamma,
                         const float* __restrict__ beta, float invM) {
  size_t idx = (size_t)blockIdx.x * blockDim.x + threadIdx.x;
  if (idx >= (size_t)M * O) return;
  int c = (int)(idx % O);
  float mean = stats[c] * invM;
  float var  = stats[O + c] * invM - mean * mean;
  float v = (float)x[idx];
  v = (v - mean) * rsqrtf(var + 1e-5f) * gamma[c] + beta[c];
  x[idx] = (_Float16)fmaxf(v, 0.f);
}

// ---------------------------------------------------------------------------
// Farthest point sampling: one block per batch, dists in registers,
// LDS argmax tree (first-index tie-break to match jnp.argmax).
// ---------------------------------------------------------------------------
__global__ void fps_kernel(const float* __restrict__ xyz, int n, int npoint,
                           int* __restrict__ outIdx) {
  __shared__ float rv[256];
  __shared__ int   ri[256];
  const int b = blockIdx.x;
  const float* base = xyz + (size_t)b * n * 3;
  float dl[32];
  const int cnt = (n + 255) / 256;
  for (int j = 0; j < cnt; ++j) dl[j] = 1e10f;
  int last = 0;
  if (threadIdx.x == 0) outIdx[(size_t)b * npoint] = 0;
  for (int it = 1; it < npoint; ++it) {
    float px = base[3 * last], py = base[3 * last + 1], pz = base[3 * last + 2];
    float best = -1.f; int bi = n;
    for (int j = 0; j < cnt; ++j) {
      int i = threadIdx.x + j * 256;
      if (i < n) {
        float dx = base[3 * i] - px, dy = base[3 * i + 1] - py,
              dz = base[3 * i + 2] - pz;
        float d = dx * dx + dy * dy + dz * dz;
        float dm = fminf(dl[j], d);
        dl[j] = dm;
        if (dm > best) { best = dm; bi = i; }
      }
    }
    rv[threadIdx.x] = best; ri[threadIdx.x] = bi;
    __syncthreads();
    for (int s = 128; s > 0; s >>= 1) {
      if (threadIdx.x < s) {
        float o = rv[threadIdx.x + s];
        if (o > rv[threadIdx.x] ||
            (o == rv[threadIdx.x] && ri[threadIdx.x + s] < ri[threadIdx.x])) {
          rv[threadIdx.x] = o; ri[threadIdx.x] = ri[threadIdx.x + s];
        }
      }
      __syncthreads();
    }
    last = ri[0];
    if (threadIdx.x == 0) outIdx[(size_t)b * npoint + it] = last;
    __syncthreads();
  }
}

__global__ void gather3(const float* __restrict__ src, const int* __restrict__ idx,
                        int S, int n, int B, float* __restrict__ dst) {
  int t = blockIdx.x * blockDim.x + threadIdx.x;
  if (t >= B * S) return;
  int b = t / S;
  int i = idx[t];
  const float* s = src + ((size_t)b * n + i) * 3;
  float* d = dst + (size_t)t * 3;
  d[0] = s[0]; d[1] = s[1]; d[2] = s[2];
}

// Ball query: first nsample in-ball indices in ascending order; empty -> n-1;
// short -> pad with first index. One thread per center.
__global__ void ball_query(const float* __restrict__ xyz,
                           const float* __restrict__ centers,
                           int n, int S, int B, float r2, int* __restrict__ g) {
  int t = blockIdx.x * blockDim.x + threadIdx.x;
  if (t >= B * S) return;
  int b = t / S;
  const float* c = centers + (size_t)t * 3;
  float cx = c[0], cy = c[1], cz = c[2];
  const float* p = xyz + (size_t)b * n * 3;
  int* out = g + (size_t)t * 64;
  int cnt = 0;
  for (int i = 0; i < n && cnt < 64; ++i) {
    float dx = p[3 * i] - cx, dy = p[3 * i + 1] - cy, dz = p[3 * i + 2] - cz;
    if (dx * dx + dy * dy + dz * dz <= r2) out[cnt++] = i;
  }
  int pad = (cnt == 0) ? (n - 1) : out[0];
  for (int s2 = cnt; s2 < 64; ++s2) out[s2] = pad;
}

// ---------------------------------------------------------------------------
// SA input builders (rows = (b,s,k), padded-K f16).
// ---------------------------------------------------------------------------
__global__ void build_sa1(const float* __restrict__ pc, const float* __restrict__ nrm,
                          const int* __restrict__ g1, const float* __restrict__ xyz1,
                          _Float16* __restrict__ out) {
  int m = blockIdx.x * blockDim.x + threadIdx.x;
  if (m >= BATCH * 512 * 64) return;
  int s = (m >> 6) & 511, b = m >> 15;
  int gi = g1[m];
  const float* ctr = xyz1 + (size_t)(b * 512 + s) * 3;
  const float* p = pc  + ((size_t)b * NPTS + gi) * 3;
  const float* nn = nrm + ((size_t)b * NPTS + gi) * 3;
  _Float16* o = out + (size_t)m * 32;
  o[0] = (_Float16)(p[0] - ctr[0]);
  o[1] = (_Float16)(p[1] - ctr[1]);
  o[2] = (_Float16)(p[2] - ctr[2]);
  o[3] = (_Float16)nn[0]; o[4] = (_Float16)nn[1]; o[5] = (_Float16)nn[2];
  for (int c = 6; c < 32; ++c) o[c] = (_Float16)0.f;
}

__global__ void build_sa2(const float* __restrict__ xyz1, const float* __restrict__ nrm1,
                          const float* __restrict__ feat1, const int* __restrict__ g2,
                          const float* __restrict__ xyz2, _Float16* __restrict__ out) {
  int m = blockIdx.x * blockDim.x + threadIdx.x;
  if (m >= BATCH * 128 * 64) return;
  int s = (m >> 6) & 127, b = m >> 13;
  int gi = g2[m];
  const float* ctr = xyz2 + (size_t)(b * 128 + s) * 3;
  const float* p  = xyz1 + ((size_t)b * 512 + gi) * 3;
  const float* nn = nrm1 + ((size_t)b * 512 + gi) * 3;
  const float* f  = feat1 + (size_t)b * 131 * 512 + gi;
  _Float16* o = out + (size_t)m * 160;
  o[0] = (_Float16)(p[0] - ctr[0]);
  o[1] = (_Float16)(p[1] - ctr[1]);
  o[2] = (_Float16)(p[2] - ctr[2]);
  o[3] = (_Float16)nn[0]; o[4] = (_Float16)nn[1]; o[5] = (_Float16)nn[2];
  for (int c = 0; c < 131; ++c) o[6 + c] = (_Float16)f[(size_t)c * 512];
  for (int c = 137; c < 160; ++c) o[c] = (_Float16)0.f;
}

__global__ void nrm_to_feat1(const float* __restrict__ nrm1, float* __restrict__ feat1) {
  int t = blockIdx.x * blockDim.x + threadIdx.x;
  if (t >= BATCH * 512 * 3) return;
  int b = t / (512 * 3), r = t % (512 * 3), s = r / 3, c = r % 3;
  feat1[(size_t)b * 131 * 512 + (size_t)c * 512 + s] = nrm1[t];
}

// Max over k dimension of f16 rows -> f32 (B, C, S) with channel offset.
__global__ void maxpool_k(const _Float16* __restrict__ rows, int ld, int S, int K,
                          int C, float* __restrict__ out, int outB, int outC,
                          int coff, int B) {
  int t = blockIdx.x * blockDim.x + threadIdx.x;
  if (t >= B * S * C) return;
  int c = t % C, bs = t / C, b = bs / S, s = bs % S;
  const _Float16* base = rows + (size_t)bs * K * ld + c;
  float m = -1e30f;
  for (int k = 0; k < K; ++k) m = fmaxf(m, (float)base[(size_t)k * ld]);
  out[(size_t)b * outB + (size_t)(c + coff) * outC + s] = m;
}

// Pack up to two (B,C,Np) f32 sources (with strides; ns=0 => broadcast over n)
// into f16 rows [b*Np+n][ld], zero pad.
__global__ void pack_rows(const float* __restrict__ p1, long long bs1, long long cs1,
                          long long ns1, int C1,
                          const float* __restrict__ p2, long long bs2, long long cs2,
                          long long ns2, int C2,
                          int Np, int ld, int M, _Float16* __restrict__ out) {
  size_t idx = (size_t)blockIdx.x * blockDim.x + threadIdx.x;
  if (idx >= (size_t)M * ld) return;
  int m = (int)(idx / ld), col = (int)(idx % ld);
  int b = m / Np, nn = m % Np;
  float v = 0.f;
  if (col < C1)            v = p1[b * bs1 + col * cs1 + nn * ns1];
  else if (col < C1 + C2)  v = p2[b * bs2 + (long long)(col - C1) * cs2 + nn * ns2];
  out[idx] = (_Float16)v;
}

__global__ void rows_to_bcn(const _Float16* __restrict__ rows, int ld, int C,
                            int Np, float* __restrict__ out) {
  size_t t = (size_t)blockIdx.x * blockDim.x + threadIdx.x;
  if (t >= (size_t)BATCH * C * Np) return;
  int b = (int)(t / ((size_t)C * Np));
  int r = (int)(t % ((size_t)C * Np));
  int c = r / Np, nn = r % Np;
  out[t] = (float)rows[((size_t)b * Np + nn) * ld + c];
}

// 3-NN (smallest d2, first-occurrence tie-break) + inverse-distance weights.
__global__ void knn3(const float* __restrict__ unk, const float* __restrict__ kn,
                     int U, int Kn, int B, int* __restrict__ oidx,
                     float* __restrict__ ow) {
  int t = blockIdx.x * blockDim.x + threadIdx.x;
  if (t >= B * U) return;
  int b = t / U;
  const float* q = unk + (size_t)t * 3;
  const float* p = kn + (size_t)b * Kn * 3;
  float qx = q[0], qy = q[1], qz = q[2];
  float d0 = 1e30f, d1 = 1e30f, d2v = 1e30f;
  int i0 = 0, i1 = 0, i2 = 0;
  for (int i = 0; i < Kn; ++i) {
    float dx = qx - p[3 * i], dy = qy - p[3 * i + 1], dz = qz - p[3 * i + 2];
    float d = dx * dx + dy * dy + dz * dz;
    if (d < d0)      { d2v = d1; i2 = i1; d1 = d0; i1 = i0; d0 = d; i0 = i; }
    else if (d < d1) { d2v = d1; i2 = i1; d1 = d; i1 = i; }
    else if (d < d2v){ d2v = d; i2 = i; }
  }
  float w0 = 1.f / (d0 + 1e-8f), w1 = 1.f / (d1 + 1e-8f), w2 = 1.f / (d2v + 1e-8f);
  float s = w0 + w1 + w2;
  oidx[t * 3] = i0; oidx[t * 3 + 1] = i1; oidx[t * 3 + 2] = i2;
  ow[t * 3] = w0 / s; ow[t * 3 + 1] = w1 / s; ow[t * 3 + 2] = w2 / s;
}

// Interpolated upsample: out32 = f32 (B,C,U) OR out16 = f16 rows [b*U+u][ld].
__global__ void interp_apply(const float* __restrict__ feats, int C, int Kn,
                             int U, int B, const int* __restrict__ oidx,
                             const float* __restrict__ ow,
                             float* __restrict__ out32,
                             _Float16* __restrict__ out16, int ld) {
  size_t t = (size_t)blockIdx.x * blockDim.x + threadIdx.x;
  if (t >= (size_t)B * U * C) return;
  int c = (int)(t % C);
  int u = (int)((t / C) % U);
  int b = (int)(t / ((size_t)C * U));
  size_t bu = (size_t)b * U + u;
  int i0 = oidx[bu * 3], i1 = oidx[bu * 3 + 1], i2 = oidx[bu * 3 + 2];
  float w0 = ow[bu * 3], w1 = ow[bu * 3 + 1], w2 = ow[bu * 3 + 2];
  const float* f = feats + (size_t)b * C * Kn + (size_t)c * Kn;
  float v = w0 * f[i0] + w1 * f[i1] + w2 * f[i2];
  if (out32) out32[(size_t)b * C * U + (size_t)c * U + u] = v;
  else       out16[bu * ld + c] = (_Float16)v;
}

// ---------------------------------------------------------------------------
// Host driver.
// ---------------------------------------------------------------------------
extern "C" void kernel_launch(void* const* d_in, const int* in_sizes, int n_in,
                              void* d_out, int out_size, void* d_ws, size_t ws_size,
                              hipStream_t stream) {
  (void)in_sizes; (void)n_in; (void)out_size; (void)ws_size;
  const float* pc     = (const float*)d_in[0];
  const float* normal = (const float*)d_in[1];
  // d_in[2] = cls (unused by the reference forward)
  auto F = [&](int i) { return (const float*)d_in[i]; };

  // Workspace arena (bump allocator, 256B aligned).
  char* ws = (char*)d_ws; size_t off = 0;
  auto alloc = [&](size_t bytes) -> void* {
    void* p = ws + off; off = (off + bytes + 255) & ~(size_t)255; return p;
  };
  _Float16* big0 = (_Float16*)alloc((size_t)524288 * 64 * 2);    // 67 MB
  _Float16* big1 = (_Float16*)alloc((size_t)524288 * 128 * 2);   // 134 MB
  float* stats = (float*)alloc(2 * 1024 * 4);
  int*   idx1  = (int*)alloc(BATCH * 512 * 4);
  int*   idx2  = (int*)alloc(BATCH * 128 * 4);
  float* xyz1  = (float*)alloc(BATCH * 512 * 3 * 4);
  float* nrm1  = (float*)alloc(BATCH * 512 * 3 * 4);
  float* xyz2  = (float*)alloc(BATCH * 128 * 3 * 4);
  int*   g1    = (int*)alloc((size_t)BATCH * 512 * 64 * 4);
  int*   g2    = (int*)alloc((size_t)BATCH * 128 * 64 * 4);
  float* feat1 = (float*)alloc((size_t)BATCH * 131 * 512 * 4);
  float* feat2 = (float*)alloc((size_t)BATCH * 256 * 128 * 4);
  float* feat3 = (float*)alloc((size_t)BATCH * 1024 * 4);
  float* f2    = (float*)alloc((size_t)BATCH * 256 * 128 * 4);
  float* up1   = (float*)alloc((size_t)BATCH * 256 * 512 * 4);
  float* f1o   = (float*)alloc((size_t)BATCH * 128 * 512 * 4);
  int*   knn1i = (int*)alloc((size_t)BATCH * 512 * 3 * 4);
  float* knn1w = (float*)alloc((size_t)BATCH * 512 * 3 * 4);
  int*   knn0i = (int*)alloc((size_t)BATCH * NPTS * 3 * 4);
  float* knn0w = (float*)alloc((size_t)BATCH * NPTS * 3 * 4);

  // Param layout: dict insertion order; each conv_bn layer = (W, b, gamma, beta).
  struct LW { const float *b, *g, *be; int O, Kp; _Float16* w16; };
  auto mk = [&](int base, int O, int K, int Op) -> LW {
    LW L; L.b = F(base + 1); L.g = F(base + 2); L.be = F(base + 3);
    L.O = O; L.Kp = (K + 31) & ~31;
    L.w16 = (_Float16*)alloc((size_t)Op * L.Kp * 2);
    int tot = Op * L.Kp;
    convert_w<<<(tot + 255) / 256, 256, 0, stream>>>(F(base), O, K, L.w16, Op, L.Kp);
    return L;
  };
  LW sa1L[3] = { mk(3, 64, 6, 64),      mk(7, 64, 64, 64),    mk(11, 128, 64, 128) };
  LW sa2L[3] = { mk(15, 128, 137, 128), mk(19, 128, 128, 128), mk(23, 256, 128, 256) };
  LW sa3L[3] = { mk(27, 256, 256, 256), mk(31, 512, 256, 512), mk(35, 1024, 512, 1024) };
  LW fp2L[2] = { mk(39, 256, 1280, 256), mk(43, 256, 256, 256) };
  LW fp1L[2] = { mk(47, 256, 384, 256),  mk(51, 128, 256, 128) };
  LW fp0L[3] = { mk(55, 128, 128, 128), mk(59, 128, 128, 128), mk(63, 128, 128, 128) };
  LW fc1L    =   mk(67, 128, 128, 128);
  _Float16* wfc2 = (_Float16*)alloc((size_t)64 * 128 * 2);
  convert_w<<<(64 * 128 + 255) / 256, 256, 0, stream>>>(F(71), 50, 128, wfc2, 64, 128);

  // GEMM + batch-stat BN + ReLU (f16 in-place).
  auto layer = [&](const LW& L, const _Float16* in, _Float16* out, int M) {
    dim3 g(M / 64, (L.O + 63) / 64);
    wmma_gemm<<<g, 128, 0, stream>>>(in, L.Kp, L.w16, L.b, M, L.O, L.Kp,
                                     out, L.O, (float*)nullptr, 0);
    hipMemsetAsync(stats, 0, (size_t)2 * L.O * 4, stream);
    dim3 gs(L.O, (M + 4095) / 4096);
    bn_stats<<<gs, 256, 0, stream>>>(out, L.O, M, stats);
    size_t tot = (size_t)M * L.O;
    bn_apply<<<(tot + 255) / 256, 256, 0, stream>>>(out, M, L.O, stats, L.g, L.be,
                                                    1.0f / (float)M);
  };

  // ---- Set abstraction 1 ----
  fps_kernel<<<BATCH, 256, 0, stream>>>(pc, NPTS, 512, idx1);
  gather3<<<(BATCH * 512 + 255) / 256, 256, 0, stream>>>(pc, idx1, 512, NPTS, BATCH, xyz1);
  gather3<<<(BATCH * 512 + 255) / 256, 256, 0, stream>>>(normal, idx1, 512, NPTS, BATCH, nrm1);
  ball_query<<<(BATCH * 512 + 255) / 256, 256, 0, stream>>>(pc, xyz1, NPTS, 512, BATCH, 0.04f, g1);
  build_sa1<<<(BATCH * 512 * 64 + 255) / 256, 256, 0, stream>>>(pc, normal, g1, xyz1, big0);
  layer(sa1L[0], big0, big1, 524288);
  layer(sa1L[1], big1, big0, 524288);
  layer(sa1L[2], big0, big1, 524288);
  maxpool_k<<<(BATCH * 512 * 128 + 255) / 256, 256, 0, stream>>>(
      big1, 128, 512, 64, 128, feat1, 131 * 512, 512, 3, BATCH);
  nrm_to_feat1<<<(BATCH * 512 * 3 + 255) / 256, 256, 0, stream>>>(nrm1, feat1);

  // ---- Set abstraction 2 ----
  fps_kernel<<<BATCH, 256, 0, stream>>>(xyz1, 512, 128, idx2);
  gather3<<<(BATCH * 128 + 255) / 256, 256, 0, stream>>>(xyz1, idx2, 128, 512, BATCH, xyz2);
  ball_query<<<(BATCH * 128 + 255) / 256, 256, 0, stream>>>(xyz1, xyz2, 512, 128, BATCH, 0.16f, g2);
  build_sa2<<<(BATCH * 128 * 64 + 255) / 256, 256, 0, stream>>>(xyz1, nrm1, feat1, g2, xyz2, big0);
  layer(sa2L[0], big0, big1, 131072);
  layer(sa2L[1], big1, big0, 131072);
  layer(sa2L[2], big0, big1, 131072);
  maxpool_k<<<(BATCH * 128 * 256 + 255) / 256, 256, 0, stream>>>(
      big1, 256, 128, 64, 256, feat2, 256 * 128, 128, 0, BATCH);

  // ---- Set abstraction 3 (global) ----
  pack_rows<<<((size_t)2048 * 256 + 255) / 256, 256, 0, stream>>>(
      feat2, 256 * 128, 128, 1, 256, (const float*)nullptr, 0, 0, 0, 0,
      128, 256, 2048, big0);
  layer(sa3L[0], big0, big1, 2048);
  layer(sa3L[1], big1, big0, 2048);
  layer(sa3L[2], big0, big1, 2048);
  maxpool_k<<<(BATCH * 1024 + 255) / 256, 256, 0, stream>>>(
      big1, 1024, 1, 128, 1024, feat3, 1024, 1, 0, BATCH);

  // ---- FP2 ----
  pack_rows<<<((size_t)2048 * 1280 + 255) / 256, 256, 0, stream>>>(
      feat3, 1024, 1, 0, 1024, feat2, 256 * 128, 128, 1, 256,
      128, 1280, 2048, big0);
  layer(fp2L[0], big0, big1, 2048);
  layer(fp2L[1], big1, big0, 2048);
  rows_to_bcn<<<(BATCH * 256 * 128 + 255) / 256, 256, 0, stream>>>(big0, 256, 256, 128, f2);

  // ---- Interp 128 -> 512, FP1 ----
  knn3<<<(BATCH * 512 + 255) / 256, 256, 0, stream>>>(xyz1, xyz2, 512, 128, BATCH, knn1i, knn1w);
  interp_apply<<<((size_t)BATCH * 512 * 256 + 255) / 256, 256, 0, stream>>>(
      f2, 256, 128, 512, BATCH, knn1i, knn1w, up1, (_Float16*)nullptr, 0);
  pack_rows<<<((size_t)8192 * 384 + 255) / 256, 256, 0, stream>>>(
      up1, 256 * 512, 512, 1, 256, feat1 + 3 * 512, 131 * 512, 512, 1, 128,
      512, 384, 8192, big0);
  layer(fp1L[0], big0, big1, 8192);
  layer(fp1L[1], big1, big0, 8192);
  rows_to_bcn<<<(BATCH * 128 * 512 + 255) / 256, 256, 0, stream>>>(big0, 128, 128, 512, f1o);

  // ---- Interp 512 -> 8192, FP0 + head ----
  knn3<<<(BATCH * NPTS + 255) / 256, 256, 0, stream>>>(pc, xyz1, NPTS, 512, BATCH, knn0i, knn0w);
  interp_apply<<<((size_t)BATCH * NPTS * 128 + 255) / 256, 256, 0, stream>>>(
      f1o, 128, 512, NPTS, BATCH, knn0i, knn0w, (float*)nullptr, big1, 128);
  layer(fp0L[0], big1, big0, 131072);
  layer(fp0L[1], big0, big1, 131072);
  layer(fp0L[2], big1, big0, 131072);
  layer(fc1L,    big0, big1, 131072);
  // Final classifier 128 -> 50, f32 straight to d_out (B, N, 50).
  dim3 gf(131072 / 64, 1);
  wmma_gemm<<<gf, 128, 0, stream>>>(big1, 128, wfc2, F(72), 131072, 50, 128,
                                    (_Float16*)nullptr, 0, (float*)d_out, 50);
}
